// Convolution_1288490189205
// MI455X (gfx1250) — compile-verified, matrix operand order checked by reading, not run on hardware
//
#include <hip/hip_runtime.h>
#include <hip/hip_bf16.h>

// ---------------------------------------------------------------------------
// Equivariant convolution (e3nn-style) for MI455X / gfx1250.
// bf16 WMMA (16x16x32) for the per-edge radial MLP with LDS-staged bf16
// weights; f32 WMMA (16x16x4) for node-side GEMMs; messages consumed straight
// from WMMA accumulator fragments; L2-resident f32 atomics for segment sums.
// ---------------------------------------------------------------------------

#define NNODES   50000
#define NEDGES   800000
#define NBASIS   10
#define NRADIAL  100
#define WNUMEL   192

#define INV_SQRT10   0.31622776601683794f
#define RELU_GAIN    1.41421356237309515f
#define INV_SQRT3    0.57735026918962576f
#define MSCALE       0.025f                   // (1/sqrt(100)) * (1/sqrt(16))
#define C_SI0        0.125f                   // 1/sqrt(64)
#define C_SI1        0.17677669529663689f     // 1/sqrt(32)
#define C_L2         0.05103103630798288f     // 0.5/sqrt(96)

typedef float  v2f     __attribute__((ext_vector_type(2)));
typedef float  v8f     __attribute__((ext_vector_type(8)));
typedef __bf16 bf16x16 __attribute__((ext_vector_type(16)));

union BFrag {            // bf16x16 viewed as 8 packed dwords (VGPR pairs)
  bf16x16 v;
  unsigned int u[8];
};

__device__ __forceinline__ v8f wmma_f32x4(v2f a, v2f b, v8f c) {
  return __builtin_amdgcn_wmma_f32_16x16x4_f32(false, a, false, b, (short)0, c,
                                               false, false);
}
__device__ __forceinline__ v8f wmma_bf16k32(bf16x16 a, bf16x16 b, v8f c) {
  return __builtin_amdgcn_wmma_f32_16x16x32_bf16(false, a, false, b, (short)0,
                                                 c, false, false);
}
__device__ __forceinline__ __bf16 tobf(float f) { return (__bf16)f; }

// ---------------------------------------------------------------------------
// Kernel A: l0 = x0@W_l1_0 * a/8 ; l1[:, :, d] = x1d@W_l1_1 * a/sqrt(32)
// One wave per 16-node tile, f32 WMMA.
// ---------------------------------------------------------------------------
__global__ __launch_bounds__(32) void node_l_kernel(
    const float* __restrict__ ni, const float* __restrict__ na,
    const float* __restrict__ Wl10, const float* __restrict__ Wl11,
    float* __restrict__ l0, float* __restrict__ l1)
{
  const int lane = threadIdx.x & 31;
  const int m = lane & 15, hi = lane >> 4;
  const int n0 = blockIdx.x * 16;
  const float* xrow = ni + (n0 + m) * 160;

  v8f acc[4] = {{}, {}, {}, {}};
  for (int k = 0; k < 64; k += 4) {
    v2f a;
    a.x = xrow[k + 2 * hi];
    a.y = xrow[k + 2 * hi + 1];
#pragma unroll
    for (int nt = 0; nt < 4; ++nt) {
      v2f b;
      b.x = Wl10[(k + 2 * hi) * 64 + nt * 16 + m];
      b.y = Wl10[(k + 2 * hi + 1) * 64 + nt * 16 + m];
      acc[nt] = wmma_f32x4(a, b, acc[nt]);
    }
  }
#pragma unroll
  for (int nt = 0; nt < 4; ++nt)
#pragma unroll
    for (int v = 0; v < 8; ++v) {
      int row = n0 + v + 8 * hi;
      l0[row * 64 + nt * 16 + m] = acc[nt][v] * na[row] * C_SI0;
    }

#pragma unroll
  for (int d = 0; d < 3; ++d) {
    v8f a1[2] = {{}, {}};
    for (int k = 0; k < 32; k += 4) {
      v2f a;
      a.x = xrow[64 + (k + 2 * hi) * 3 + d];
      a.y = xrow[64 + (k + 2 * hi + 1) * 3 + d];
#pragma unroll
      for (int nt = 0; nt < 2; ++nt) {
        v2f b;
        b.x = Wl11[(k + 2 * hi) * 32 + nt * 16 + m];
        b.y = Wl11[(k + 2 * hi + 1) * 32 + nt * 16 + m];
        a1[nt] = wmma_f32x4(a, b, a1[nt]);
      }
    }
#pragma unroll
    for (int nt = 0; nt < 2; ++nt)
#pragma unroll
      for (int v = 0; v < 8; ++v) {
        int row = n0 + v + 8 * hi;
        l1[row * 96 + (nt * 16 + m) * 3 + d] = a1[nt][v] * na[row] * C_SI1;
      }
  }
}

// ---------------------------------------------------------------------------
// Kernel B: per-edge radial MLP (bf16 WMMA, weights bf16-transposed in LDS) +
// messages formed directly from accumulator fragments + atomic scatter.
// 4 waves/block, each wave owns a 16-edge tile.
// ---------------------------------------------------------------------------
#define EDGE_WAVES 4

__global__ __launch_bounds__(128) void edge_kernel(
    const float* __restrict__ ele, const float* __restrict__ eattr,
    const int* __restrict__ esrc, const int* __restrict__ edst,
    const float* __restrict__ Wfc1, const float* __restrict__ Wfc2,
    const float* __restrict__ l0, const float* __restrict__ l1,
    float* __restrict__ s0, float* __restrict__ s1,
    float* __restrict__ s2, float* __restrict__ s3)
{
  // Transposed bf16 weights: [n][k], k padded so (k0,k0+1) pairs are one dword.
  __shared__ __align__(16) __bf16 sWfc1T[NRADIAL * 12];    //  2.4 KB
  __shared__ __align__(16) __bf16 sWfc2T[WNUMEL * 104];    // 39.9 KB
  __shared__ __align__(16) __bf16 sH[EDGE_WAVES][16 * 104];// 13.3 KB

  const int tid = threadIdx.x;
  for (int n = tid; n < NRADIAL; n += 128)
#pragma unroll
    for (int k = 0; k < 12; ++k)
      sWfc1T[n * 12 + k] = (k < NBASIS) ? tobf(Wfc1[k * NRADIAL + n]) : tobf(0.f);
  for (int n = tid; n < WNUMEL; n += 128)
    for (int k = 0; k < 104; ++k)
      sWfc2T[n * 104 + k] = (k < NRADIAL) ? tobf(Wfc2[k * WNUMEL + n]) : tobf(0.f);
  __syncthreads();

  const int wv = tid >> 5, lane = tid & 31;
  const int m = lane & 15, hi = lane >> 4;
  const int e0 = (blockIdx.x * EDGE_WAVES + wv) * 16;

  // Prefetch gathered source rows into cache while we run the MLP.
  if (lane < 16) {
    int s = esrc[e0 + lane];
    __builtin_prefetch(l0 + s * 64, 0, 0);
    __builtin_prefetch(l1 + s * 96, 0, 0);
  }

  // ---- GEMM1: H(16x100) = ELE(16x10) @ Wfc1, relu, gain --------------------
  {
    BFrag af;
    const float2* ap = (const float2*)(ele + (e0 + m) * NBASIS);
#pragma unroll
    for (int v = 0; v < 8; ++v) {
      int k0 = (v >> 2) * 16 + hi * 8 + (v & 3) * 2;
      float x0 = 0.f, x1 = 0.f;
      if (k0 < NBASIS) { float2 p = ap[k0 >> 1]; x0 = p.x; x1 = p.y; }
      af.v[2 * v] = tobf(x0);
      af.v[2 * v + 1] = tobf(x1);
    }
    v8f hacc[7];
#pragma unroll
    for (int nt = 0; nt < 7; ++nt) {
      BFrag bf;
      int n = nt * 16 + m;
      const unsigned int* bp = (const unsigned int*)(sWfc1T + n * 12);
#pragma unroll
      for (int v = 0; v < 8; ++v) {
        int k0 = (v >> 2) * 16 + hi * 8 + (v & 3) * 2;
        bf.u[v] = (n < NRADIAL && k0 < NBASIS) ? bp[k0 >> 1] : 0u;
      }
      v8f z = {};
      hacc[nt] = wmma_bf16k32(af.v, bf.v, z);
    }
    // relu(x/sqrt(10))*sqrt(2) -> LDS (bf16, same precision GEMM2 consumes)
#pragma unroll
    for (int nt = 0; nt < 7; ++nt) {
      int col = nt * 16 + m;
      if (col < NRADIAL) {
#pragma unroll
        for (int v = 0; v < 8; ++v) {
          float x = hacc[nt][v] * INV_SQRT10;
          sH[wv][(v + 8 * hi) * 104 + col] = tobf(fmaxf(x, 0.f) * RELU_GAIN);
        }
      }
    }
  }

  // ---- GEMM2: W(16x192) = H(16x100) @ Wfc2 ---------------------------------
  v8f wacc[12];
#pragma unroll
  for (int nt = 0; nt < 12; ++nt) wacc[nt] = (v8f){};
#pragma unroll
  for (int kk = 0; kk < 4; ++kk) {
    BFrag af;
    const unsigned int* apk = (const unsigned int*)(sH[wv] + m * 104);
#pragma unroll
    for (int v = 0; v < 8; ++v) {
      int k0 = kk * 32 + (v >> 2) * 16 + hi * 8 + (v & 3) * 2;
      af.u[v] = (k0 < NRADIAL) ? apk[k0 >> 1] : 0u;
    }
#pragma unroll
    for (int nt = 0; nt < 12; ++nt) {
      BFrag bf;
      const unsigned int* bp = (const unsigned int*)(sWfc2T + (nt * 16 + m) * 104);
#pragma unroll
      for (int v = 0; v < 8; ++v) {
        int k0 = kk * 32 + (v >> 2) * 16 + hi * 8 + (v & 3) * 2;
        bf.u[v] = (k0 < NRADIAL) ? bp[k0 >> 1] : 0u;
      }
      wacc[nt] = wmma_bf16k32(af.v, bf.v, wacc[nt]);
    }
  }

  // ---- Messages straight from fragments + atomic segment scatter -----------
  // Fragment element (lane, nt, v) <-> edge e = v + 8*hi, channel u = nt*16+m.
  int srcv[8], dstv[8];
  float ea0v[8], e1xv[8], e1yv[8], e1zv[8];
#pragma unroll
  for (int v = 0; v < 8; ++v) {
    int eg = e0 + v + 8 * hi;
    srcv[v] = esrc[eg];
    dstv[v] = edst[eg];
    float4 ea = *(const float4*)(eattr + eg * 4);
    ea0v[v] = ea.x; e1xv[v] = ea.y; e1yv[v] = ea.z; e1zv[v] = ea.w;
  }

  // m0 (w cols 0..63) and m1 (w cols 64..127) share the gathered y0.
#pragma unroll
  for (int nt = 0; nt < 4; ++nt) {
    int u = nt * 16 + m;
#pragma unroll
    for (int v = 0; v < 8; ++v) {
      float w0v = wacc[nt][v] * MSCALE;
      float w1v = wacc[nt + 4][v] * MSCALE;
      float y0 = l0[srcv[v] * 64 + u];
      atomicAdd(&s0[dstv[v] * 64 + u], w0v * y0 * ea0v[v]);
      float b = w1v * y0;
      float* s1p = s1 + dstv[v] * 192 + u * 3;
      atomicAdd(s1p + 0, b * e1xv[v]);
      atomicAdd(s1p + 1, b * e1yv[v]);
      atomicAdd(s1p + 2, b * e1zv[v]);
    }
  }
  // m2 (w cols 128..159) and m3 (w cols 160..191) share the gathered y1.
#pragma unroll
  for (int uh = 0; uh < 2; ++uh) {
    int u = uh * 16 + m;
#pragma unroll
    for (int v = 0; v < 8; ++v) {
      float w2v = wacc[8 + uh][v] * MSCALE;
      float w3v = wacc[10 + uh][v] * (MSCALE * INV_SQRT3);
      const float* y1p = l1 + srcv[v] * 96 + u * 3;
      float y1x = y1p[0], y1y = y1p[1], y1z = y1p[2];
      float c = w2v * ea0v[v];
      float* s2p = s2 + dstv[v] * 96 + u * 3;
      atomicAdd(s2p + 0, c * y1x);
      atomicAdd(s2p + 1, c * y1y);
      atomicAdd(s2p + 2, c * y1z);
      float dot = y1x * e1xv[v] + y1y * e1yv[v] + y1z * e1zv[v];
      atomicAdd(&s3[dstv[v] * 32 + u], w3v * dot);
    }
  }
}

// ---------------------------------------------------------------------------
// Kernel C: out0 = a*[ x0/8 @ Wsi0 + c2*(s0@W00 + s3@W10) ]
//           out1d = a*[ x1d/sqrt32 @ Wsi1 + c2*(s1d@W01 + s2d@W11) ]
// All terms fused into one f32-WMMA accumulator (scales folded into A).
// ---------------------------------------------------------------------------
__global__ __launch_bounds__(32) void node_out_kernel(
    const float* __restrict__ ni, const float* __restrict__ na,
    const float* __restrict__ Wsi0, const float* __restrict__ Wsi1,
    const float* __restrict__ W00, const float* __restrict__ W10,
    const float* __restrict__ W01, const float* __restrict__ W11,
    const float* __restrict__ s0, const float* __restrict__ s1,
    const float* __restrict__ s2, const float* __restrict__ s3,
    float* __restrict__ out)
{
  const int lane = threadIdx.x & 31;
  const int m = lane & 15, hi = lane >> 4;
  const int n0 = blockIdx.x * 16;
  const float* xrow = ni + (n0 + m) * 160;
  const float* s0r = s0 + (n0 + m) * 64;
  const float* s1r = s1 + (n0 + m) * 192;
  const float* s2r = s2 + (n0 + m) * 96;
  const float* s3r = s3 + (n0 + m) * 32;

  v8f acc[4] = {{}, {}, {}, {}};
  for (int k = 0; k < 64; k += 4) {
    v2f a;
    a.x = xrow[k + 2 * hi] * C_SI0;
    a.y = xrow[k + 2 * hi + 1] * C_SI0;
#pragma unroll
    for (int nt = 0; nt < 4; ++nt) {
      v2f b;
      b.x = Wsi0[(k + 2 * hi) * 64 + nt * 16 + m];
      b.y = Wsi0[(k + 2 * hi + 1) * 64 + nt * 16 + m];
      acc[nt] = wmma_f32x4(a, b, acc[nt]);
    }
  }
  for (int k = 0; k < 64; k += 4) {
    v2f a;
    a.x = s0r[k + 2 * hi] * C_L2;
    a.y = s0r[k + 2 * hi + 1] * C_L2;
#pragma unroll
    for (int nt = 0; nt < 4; ++nt) {
      v2f b;
      b.x = W00[(k + 2 * hi) * 64 + nt * 16 + m];
      b.y = W00[(k + 2 * hi + 1) * 64 + nt * 16 + m];
      acc[nt] = wmma_f32x4(a, b, acc[nt]);
    }
  }
  for (int k = 0; k < 32; k += 4) {
    v2f a;
    a.x = s3r[k + 2 * hi] * C_L2;
    a.y = s3r[k + 2 * hi + 1] * C_L2;
#pragma unroll
    for (int nt = 0; nt < 4; ++nt) {
      v2f b;
      b.x = W10[(k + 2 * hi) * 64 + nt * 16 + m];
      b.y = W10[(k + 2 * hi + 1) * 64 + nt * 16 + m];
      acc[nt] = wmma_f32x4(a, b, acc[nt]);
    }
  }
#pragma unroll
  for (int nt = 0; nt < 4; ++nt)
#pragma unroll
    for (int v = 0; v < 8; ++v) {
      int row = n0 + v + 8 * hi;
      out[row * 160 + nt * 16 + m] = acc[nt][v] * na[row];
    }

#pragma unroll
  for (int d = 0; d < 3; ++d) {
    v8f a1[2] = {{}, {}};
    for (int k = 0; k < 32; k += 4) {
      v2f a;
      a.x = xrow[64 + (k + 2 * hi) * 3 + d] * C_SI1;
      a.y = xrow[64 + (k + 2 * hi + 1) * 3 + d] * C_SI1;
#pragma unroll
      for (int nt = 0; nt < 2; ++nt) {
        v2f b;
        b.x = Wsi1[(k + 2 * hi) * 32 + nt * 16 + m];
        b.y = Wsi1[(k + 2 * hi + 1) * 32 + nt * 16 + m];
        a1[nt] = wmma_f32x4(a, b, a1[nt]);
      }
    }
    for (int k = 0; k < 64; k += 4) {
      v2f a;
      a.x = s1r[(k + 2 * hi) * 3 + d] * C_L2;
      a.y = s1r[(k + 2 * hi + 1) * 3 + d] * C_L2;
#pragma unroll
      for (int nt = 0; nt < 2; ++nt) {
        v2f b;
        b.x = W01[(k + 2 * hi) * 32 + nt * 16 + m];
        b.y = W01[(k + 2 * hi + 1) * 32 + nt * 16 + m];
        a1[nt] = wmma_f32x4(a, b, a1[nt]);
      }
    }
    for (int k = 0; k < 32; k += 4) {
      v2f a;
      a.x = s2r[(k + 2 * hi) * 3 + d] * C_L2;
      a.y = s2r[(k + 2 * hi + 1) * 3 + d] * C_L2;
#pragma unroll
      for (int nt = 0; nt < 2; ++nt) {
        v2f b;
        b.x = W11[(k + 2 * hi) * 32 + nt * 16 + m];
        b.y = W11[(k + 2 * hi + 1) * 32 + nt * 16 + m];
        a1[nt] = wmma_f32x4(a, b, a1[nt]);
      }
    }
#pragma unroll
    for (int nt = 0; nt < 2; ++nt)
#pragma unroll
      for (int v = 0; v < 8; ++v) {
        int row = n0 + v + 8 * hi;
        out[row * 160 + 64 + (nt * 16 + m) * 3 + d] = a1[nt][v] * na[row];
      }
  }
}

// ---------------------------------------------------------------------------
extern "C" void kernel_launch(void* const* d_in, const int* in_sizes, int n_in,
                              void* d_out, int out_size, void* d_ws, size_t ws_size,
                              hipStream_t stream) {
  const float* node_input = (const float*)d_in[0];
  const float* node_attr  = (const float*)d_in[1];
  const float* edge_attr  = (const float*)d_in[2];
  const float* ele        = (const float*)d_in[3];
  const float* W_si0  = (const float*)d_in[4];
  const float* W_si1  = (const float*)d_in[5];
  const float* W_l1_0 = (const float*)d_in[6];
  const float* W_l1_1 = (const float*)d_in[7];
  const float* W_l2_00 = (const float*)d_in[8];
  const float* W_l2_10 = (const float*)d_in[9];
  const float* W_l2_01 = (const float*)d_in[10];
  const float* W_l2_11 = (const float*)d_in[11];
  const float* W_fc1 = (const float*)d_in[12];
  const float* W_fc2 = (const float*)d_in[13];
  const int* edge_src = (const int*)d_in[14];
  const int* edge_dst = (const int*)d_in[15];
  float* out = (float*)d_out;

  float* ws = (float*)d_ws;
  float* l0 = ws;                                 // N*64
  float* l1 = ws + (size_t)NNODES * 64;           // N*96
  float* s0 = ws + (size_t)NNODES * 160;          // N*64
  float* s1 = s0 + (size_t)NNODES * 64;           // N*192
  float* s2 = s1 + (size_t)NNODES * 192;          // N*96
  float* s3 = s2 + (size_t)NNODES * 96;           // N*32

  hipMemsetAsync(s0, 0, (size_t)NNODES * 384 * sizeof(float), stream);

  node_l_kernel<<<NNODES / 16, 32, 0, stream>>>(node_input, node_attr,
                                                W_l1_0, W_l1_1, l0, l1);

  edge_kernel<<<NEDGES / (16 * EDGE_WAVES), 128, 0, stream>>>(
      ele, edge_attr, edge_src, edge_dst, W_fc1, W_fc2, l0, l1, s0, s1, s2, s3);

  node_out_kernel<<<NNODES / 16, 32, 0, stream>>>(
      node_input, node_attr, W_si0, W_si1, W_l2_00, W_l2_10, W_l2_01, W_l2_11,
      s0, s1, s2, s3, out);
}
